// BlockTransformer_80092550136042
// MI455X (gfx1250) — compile-verified, matrix operand order checked by reading, not run on hardware
//
#include <hip/hip_runtime.h>
#include <math.h>

// ---------------- CDNA5 WMMA types ----------------
typedef __bf16 bf16;
typedef __bf16 v16bf __attribute__((ext_vector_type(16)));
typedef __bf16 v8bf  __attribute__((ext_vector_type(8)));
typedef float  v8f   __attribute__((ext_vector_type(8)));
typedef float  v4f   __attribute__((ext_vector_type(4)));
typedef int    v4i   __attribute__((ext_vector_type(4)));

__device__ inline v8f wmma_bf16(v16bf a, v16bf b, v8f c) {
  return __builtin_amdgcn_wmma_f32_16x16x32_bf16(false, a, false, b, (short)0, c, false, false);
}

// ---------------- async global->LDS copy (CDNA5 GLOBAL_LOAD_ASYNC_TO_LDS_B128) ----------------
// Probe-learned signature: param 1 is a generic `int4*` (global source); pass both as int4*.
#if __has_builtin(__builtin_amdgcn_global_load_async_to_lds_b128)
#define ASYNC_LDS_COPY 1
__device__ inline void async_cp_b128(const bf16* g, bf16* l) {
  __builtin_amdgcn_global_load_async_to_lds_b128((v4i*)g, (v4i*)l, 0, 0);
}
__device__ inline void async_wait() {
#if __has_builtin(__builtin_amdgcn_s_wait_asynccnt)
  __builtin_amdgcn_s_wait_asynccnt(0);
#else
  asm volatile("s_wait_asynccnt 0x0" ::: "memory");
#endif
}
#else
#define ASYNC_LDS_COPY 0
__device__ inline void async_cp_b128(const bf16* g, bf16* l) {
  *(v8bf*)l = *(const v8bf*)g;
}
__device__ inline void async_wait() {}
#endif

// ---------------- problem constants ----------------
constexpr int cB  = 8;
constexpr int cP  = 16;
constexpr int cH  = 32;
constexpr int cNO = 32;
constexpr int cNA = 8;
constexpr int cD  = 768;
constexpr int cL  = 12;
constexpr int cNH = 12;
constexpr int cF  = 3072;
constexpr int cDH = 64;
constexpr int cGrp = cNO + cNA;          // 40
constexpr int cN  = cP + cH * cGrp;      // 1296
constexpr float cSCALE = 0.125f;         // 1/sqrt(64)
constexpr float cNEG = -1.0e9f;

// ---------------- vectorized fragment loads (2 x b128 each) ----------------
__device__ inline v16bf ld_frag_pair(const bf16* p0, const bf16* p1) {
  v8bf lo = *(const v8bf*)p0;
  v8bf hi = *(const v8bf*)p1;
  return __builtin_shufflevector(lo, hi, 0,1,2,3,4,5,6,7,8,9,10,11,12,13,14,15);
}
// A (16x32 bf16): lane L holds row L&15; lanes 0-15 -> K 0..7 & 16..23, lanes 16-31 -> K 8..15 & 24..31
__device__ inline v16bf load_frag_a(const bf16* src, int ld) {
  int lane = threadIdx.x & 31;
  int row = lane & 15, kb = (lane >> 4) * 8;
  const bf16* r = src + row * ld;
  return ld_frag_pair(r + kb, r + 16 + kb);
}
// B (32x16 bf16) from an N-major (transposed) buffer: lane L holds col L&15; same K split.
__device__ inline v16bf load_frag_b_t(const bf16* srcT, int ld) {
  int lane = threadIdx.x & 31;
  int col = lane & 15, kb = (lane >> 4) * 8;
  const bf16* r = srcT + col * ld;
  return ld_frag_pair(r + kb, r + 16 + kb);
}

__device__ inline float gelu_tanh(float x) {
  float x3 = x * x * x;
  return 0.5f * x * (1.0f + tanhf(0.7978845608028654f * (x + 0.044715f * x3)));
}

// ---------------- token stream assembly ----------------
__global__ void k_build(const float* __restrict__ pref, const float* __restrict__ obs,
                        const float* __restrict__ act, float* __restrict__ x) {
  long idx = (long)blockIdx.x * 256 + threadIdx.x;
  long total = (long)cB * cN * cD;
  if (idx >= total) return;
  int d = (int)(idx % cD);
  int t = (int)((idx / cD) % cN);
  int b = (int)(idx / ((long)cD * cN));
  float v;
  if (t < cP) {
    v = pref[((long)b * cP + t) * cD + d];
  } else {
    int j = t - cP, h = j / cGrp, pos = j % cGrp;
    v = (pos < cNO) ? obs[(((long)b * cH + h) * cNO + pos) * cD + d]
                    : act[(((long)b * cH + h) * cNA + (pos - cNO)) * cD + d];
  }
  x[idx] = v;
}

// ---------------- LayerNorm (bf16 out for WMMA A-operands, or fp32 for final) ----------------
__global__ __launch_bounds__(256) void k_ln(const float* __restrict__ x,
                                            const float* __restrict__ s,
                                            const float* __restrict__ b,
                                            bf16* outb, float* outf, int out_fp32) {
  int row = blockIdx.x;                       // b*N + n
  const float* xr = x + (size_t)row * cD;
  __shared__ float ssum[256], ssq[256];
  float sum = 0.f, sq = 0.f;
  for (int i = threadIdx.x; i < cD; i += 256) { float v = xr[i]; sum += v; sq += v * v; }
  ssum[threadIdx.x] = sum; ssq[threadIdx.x] = sq;
  __syncthreads();
  for (int off = 128; off; off >>= 1) {
    if (threadIdx.x < off) { ssum[threadIdx.x] += ssum[threadIdx.x + off]; ssq[threadIdx.x] += ssq[threadIdx.x + off]; }
    __syncthreads();
  }
  float mu = ssum[0] / cD;
  float var = ssq[0] / cD - mu * mu;
  float inv = rsqrtf(var + 1e-6f);
  for (int i = threadIdx.x; i < cD; i += 256) {
    float v = (xr[i] - mu) * inv * s[i] + b[i];
    if (out_fp32) outf[(size_t)row * cD + i] = v;
    else          outb[(size_t)row * cD + i] = (bf16)v;
  }
}

// ---------------- weight prep: W fp32 (KxNc) -> WT bf16 (NcxK), LDS-tiled transpose ----------------
// Tile 32(K) x 64(N); coalesced f32 reads, coalesced b128 bf16 writes.
__global__ __launch_bounds__(256) void k_wprep(const float* __restrict__ W,
                                               bf16* __restrict__ WT, int K, int Nc) {
  __shared__ __align__(16) bf16 sT[32 * 72];      // padded: 144B rows
  int k0 = blockIdx.x * 32, n0 = blockIdx.y * 64;
  int t = threadIdx.x;
  {
    int r = t >> 3, c = (t & 7) * 8;              // r: k-row 0..31, c: n-col chunk
    const float* src = W + (size_t)(k0 + r) * Nc + n0 + c;
    v4f w0 = *(const v4f*)(src);
    v4f w1 = *(const v4f*)(src + 4);
    v8bf pk;
#pragma unroll
    for (int i = 0; i < 4; ++i) { pk[i] = (bf16)w0[i]; pk[4 + i] = (bf16)w1[i]; }
    *(v8bf*)(sT + r * 72 + c) = pk;
  }
  __syncthreads();
  {
    int n = t >> 2, kk = (t & 3) * 8;             // n: 0..63, kk: k chunk
    v8bf o;
#pragma unroll
    for (int i = 0; i < 8; ++i) o[i] = sT[(kk + i) * 72 + n];
    *(v8bf*)(WT + (size_t)(n0 + n) * K + k0 + kk) = o;
  }
}

// ---------------- bf16 WMMA GEMM: out = A(MxK) @ WT^T + bias  (WT is Nc x K bf16) ----------------
// Block tile 128x128, 8 waves in 4(M) x 2(N) grid; each wave: 32x64 = 2 A-frags x 4 B-frags
// -> 8 WMMAs per 12 ds_load_b128 per K-step. Staging via async global->LDS b128 copies.
// mode 0: store bf16 | mode 1: gelu -> bf16 | mode 2: residual add into fp32 outF
// mode 3: store bf16 transposed per-head: out[((b*NH+h)*DH+dd)*cN + n]  (for V)
constexpr int LDA = 40;   // padded row length (80B: 16B-aligned, 20-bank stride)
__global__ __launch_bounds__(256) void k_gemm(const bf16* __restrict__ A,
                                              const bf16* __restrict__ WT,
                                              const float* __restrict__ bias,
                                              float* outF, bf16* outB,
                                              int M, int K, int Nc, int mode) {
  __shared__ __align__(16) bf16 sA [128 * LDA];   // 128(M) x 32(K), padded
  __shared__ __align__(16) bf16 sBt[128 * LDA];   // 128(N) x 32(K), padded
  int tid = threadIdx.x, wave = tid >> 5, lane = tid & 31;
  int waveM = wave & 3, waveN = wave >> 2;
  int m0 = blockIdx.x * 128, n0 = blockIdx.y * 128;
  v8f acc[2][4];
#pragma unroll
  for (int m = 0; m < 2; ++m)
#pragma unroll
    for (int n = 0; n < 4; ++n) acc[m][n] = (v8f){0.f,0.f,0.f,0.f,0.f,0.f,0.f,0.f};

  const int rT = tid >> 1, cT = (tid & 1) * 16;   // staging: 16 bf16 per thread per tile

  for (int k0 = 0; k0 < K; k0 += 32) {
    // stage A and B tiles: 4 async b128 copies per thread (no VGPR round-trip)
    const bf16* srcA = A  + (size_t)(m0 + rT) * K + k0 + cT;
    const bf16* srcB = WT + (size_t)(n0 + rT) * K + k0 + cT;
    async_cp_b128(srcA,     sA  + rT * LDA + cT);
    async_cp_b128(srcA + 8, sA  + rT * LDA + cT + 8);
    async_cp_b128(srcB,     sBt + rT * LDA + cT);
    async_cp_b128(srcB + 8, sBt + rT * LDA + cT + 8);
    if (k0 + 32 < K)  // hint next weight tile (global_prefetch)
      __builtin_prefetch(WT + (size_t)(n0 + rT) * K + k0 + 32, 0, 1);
    async_wait();
    __syncthreads();
    // load all fragments first, then a burst of 8 WMMAs (lets XDL overlap LDS)
    v16bf a0 = load_frag_a(sA + (waveM * 32     ) * LDA, LDA);
    v16bf a1 = load_frag_a(sA + (waveM * 32 + 16) * LDA, LDA);
    v16bf bf[4];
#pragma unroll
    for (int n = 0; n < 4; ++n)
      bf[n] = load_frag_b_t(sBt + (waveN * 64 + n * 16) * LDA, LDA);
#pragma unroll
    for (int n = 0; n < 4; ++n) {
      acc[0][n] = wmma_bf16(a0, bf[n], acc[0][n]);
      acc[1][n] = wmma_bf16(a1, bf[n], acc[1][n]);
    }
    __syncthreads();
  }

  int col0 = lane & 15, hi = lane >> 4;
#pragma unroll
  for (int m = 0; m < 2; ++m) {
#pragma unroll
    for (int n = 0; n < 4; ++n) {
#pragma unroll
      for (int j = 0; j < 8; ++j) {
        int row = m0 + waveM * 32 + m * 16 + j + 8 * hi;
        int col = n0 + waveN * 64 + n * 16 + col0;
        float v = acc[m][n][j] + bias[col];
        if (mode == 0)      outB[(size_t)row * Nc + col] = (bf16)v;
        else if (mode == 1) outB[(size_t)row * Nc + col] = (bf16)gelu_tanh(v);
        else if (mode == 2) outF[(size_t)row * Nc + col] += v;
        else {              // transposed per-head store for V
          int b  = row / cN, nn = row - b * cN;
          int h  = col >> 6, dd = col & 63;
          outB[(((size_t)b * cNH + h) * cDH + dd) * cN + nn] = (bf16)v;
        }
      }
    }
  }
}

// ---------------- mask helpers (rule mask computed analytically) ----------------
__device__ inline void tok_info(int t, int& g, int& h) {
  if (t < cP) { g = 0; h = -1; }
  else { int j = t - cP; h = j / cGrp; int pos = j % cGrp; g = (pos < cNO) ? 1 : 2; }
}
__device__ inline bool rule_ok(int gq, int hq, int gk, int hk) {
  if (gk == 0) return true;
  if (gk == 1) return (gq >= 1) && (hk <= hq);
  return (gq == 2) && (hk <= hq);
}
__device__ inline bool pad_ok(const unsigned char* pm, const unsigned char* om,
                              const unsigned char* am, int b, int t) {
  if (t < cP) return pm[b * cP + t] != 0;
  int j = t - cP, h = j / cGrp, pos = j % cGrp;
  return (pos < cNO) ? (om[((size_t)b * cH + h) * cNO + pos] != 0)
                     : (am[((size_t)b * cH + h) * cNA + (pos - cNO)] != 0);
}

// ---------------- flash attention: one wave per (b, head, 16-query tile) ----------------
// q,k: (B,N,D) bf16 row-major.  vT: (B,NH,DH,N) bf16 (token-contiguous -> vector B-frags)
constexpr int LDP = 40;
__global__ __launch_bounds__(32) void k_attn(const bf16* __restrict__ q,
                                             const bf16* __restrict__ k,
                                             const bf16* __restrict__ vT,
                                             const unsigned char* __restrict__ pm,
                                             const unsigned char* __restrict__ om,
                                             const unsigned char* __restrict__ am,
                                             bf16* __restrict__ o) {
  __shared__ __align__(16) bf16 sP[16 * LDP];
  int qt = blockIdx.x, h = blockIdx.y, b = blockIdx.z;
  int lane = threadIdx.x & 31;
  int col = lane & 15, hi = lane >> 4, kb8 = hi * 8;
  size_t bb = (size_t)b * cN;
  int h64 = h * cDH;

  // Q A-fragments for the two DH chunks (2 x b128 each)
  v16bf qa[2];
  {
    const bf16* qp = q + (bb + qt * 16 + col) * cD + h64;
    qa[0] = ld_frag_pair(qp + kb8,      qp + 16 + kb8);
    qa[1] = ld_frag_pair(qp + 32 + kb8, qp + 48 + kb8);
  }

  // per-row (C-fragment row = j + 8*hi) query group/step, softmax state
  int gq[8], hq[8];
  float mrow[8], srow[8];
  v8f oacc[4];
#pragma unroll
  for (int n = 0; n < 4; ++n) oacc[n] = (v8f){0.f,0.f,0.f,0.f,0.f,0.f,0.f,0.f};
#pragma unroll
  for (int j = 0; j < 8; ++j) {
    tok_info(qt * 16 + j + 8 * hi, gq[j], hq[j]);
    mrow[j] = -INFINITY; srow[j] = 0.f;
  }

  const bf16* vTh = vT + ((size_t)b * cNH + h) * cDH * cN;

  const int nKB = (cN + 31) / 32;  // 41 (last block half-padded)
  for (int kbi = 0; kbi < nKB; ++kbi) {
    int kbase = kbi * 32;
    int keyA = kbase + col, keyB = kbase + 16 + col;   // this lane's key columns
    int keyAc = keyA < cN ? keyA : cN - 1;
    int keyBc = keyB < cN ? keyB : cN - 1;
    int gkA, hkA, gkB, hkB;
    tok_info(keyAc, gkA, hkA); tok_info(keyBc, gkB, hkB);
    bool padA = (keyA < cN) && pad_ok(pm, om, am, b, keyAc);
    bool padB = (keyB < cN) && pad_ok(pm, om, am, b, keyBc);

    // S = Q K^T for 32 keys (two 16x16 tiles, 2 WMMAs each over DH=64)
    v8f Sa = (v8f){0.f,0.f,0.f,0.f,0.f,0.f,0.f,0.f};
    v8f Sb = Sa;
    {
      const bf16* kpA = k + (bb + keyAc) * cD + h64;
      const bf16* kpB = k + (bb + keyBc) * cD + h64;
#pragma unroll
      for (int c = 0; c < 2; ++c) {
        v16bf ka  = ld_frag_pair(kpA + c * 32 + kb8, kpA + c * 32 + 16 + kb8);
        v16bf kb2 = ld_frag_pair(kpB + c * 32 + kb8, kpB + c * 32 + 16 + kb8);
        Sa = wmma_bf16(qa[c], ka,  Sa);
        Sb = wmma_bf16(qa[c], kb2, Sb);
      }
    }

    // scale + mask + online softmax (row reductions across the 16-lane half)
#pragma unroll
    for (int j = 0; j < 8; ++j) {
      float sa = Sa[j] * cSCALE;
      float sb = Sb[j] * cSCALE;
      if (!(padA && rule_ok(gq[j], hq[j], gkA, hkA))) sa = cNEG;
      if (!(padB && rule_ok(gq[j], hq[j], gkB, hkB))) sb = cNEG;
      float t = fmaxf(sa, sb);
      for (int off = 8; off; off >>= 1) t = fmaxf(t, __shfl_xor(t, off, 16));
      float mnew = fmaxf(mrow[j], t);
      float corr = expf(mrow[j] - mnew);
      srow[j] *= corr;
#pragma unroll
      for (int n = 0; n < 4; ++n) oacc[n][j] *= corr;
      float pa = expf(sa - mnew), pb = expf(sb - mnew);
      float rs = pa + pb;
      for (int off = 8; off; off >>= 1) rs += __shfl_xor(rs, off, 16);
      srow[j] += rs;
      mrow[j] = mnew;
      int prow = j + 8 * hi;                 // transpose C-layout -> LDS row-major
      sP[prow * LDP + col]      = (bf16)pa;
      sP[prow * LDP + 16 + col] = (bf16)pb;
    }
    __syncthreads();

    // O += P(16x32) @ V(32x64): B-frags are token-contiguous rows of vT (2 x b128)
    v16bf pf = load_frag_a(sP, LDP);
#pragma unroll
    for (int n = 0; n < 4; ++n) {
      const bf16* vp = vTh + (size_t)(n * 16 + col) * cN + kbase;
      v16bf vf = ld_frag_pair(vp + kb8, vp + 16 + kb8);
      oacc[n] = wmma_bf16(pf, vf, oacc[n]);
    }
    __syncthreads();
  }

  // normalize + store bf16 attention output (row-major, feeds O-projection GEMM)
#pragma unroll
  for (int n = 0; n < 4; ++n)
#pragma unroll
    for (int j = 0; j < 8; ++j) {
      int row = qt * 16 + j + 8 * hi;
      o[(bb + row) * cD + h64 + n * 16 + col] = (bf16)(oacc[n][j] / srow[j]);
    }
}

// ---------------- host orchestration ----------------
extern "C" void kernel_launch(void* const* d_in, const int* in_sizes, int n_in,
                              void* d_out, int out_size, void* d_ws, size_t ws_size,
                              hipStream_t stream) {
  (void)in_sizes; (void)n_in; (void)out_size; (void)ws_size;
  const float* prefix = (const float*)d_in[0];
  const float* obs    = (const float*)d_in[1];
  const float* act    = (const float*)d_in[2];
  const float* ln1s   = (const float*)d_in[3];
  const float* ln1b   = (const float*)d_in[4];
  const float* ln2s   = (const float*)d_in[5];
  const float* ln2b   = (const float*)d_in[6];
  const float* wq     = (const float*)d_in[7];
  const float* wk     = (const float*)d_in[8];
  const float* wv     = (const float*)d_in[9];
  const float* wo     = (const float*)d_in[10];
  const float* bq     = (const float*)d_in[11];
  const float* bk     = (const float*)d_in[12];
  const float* bv     = (const float*)d_in[13];
  const float* bo     = (const float*)d_in[14];
  const float* w1     = (const float*)d_in[15];
  const float* b1     = (const float*)d_in[16];
  const float* w2     = (const float*)d_in[17];
  const float* b2     = (const float*)d_in[18];
  const float* lnfs   = (const float*)d_in[19];
  const float* lnfb   = (const float*)d_in[20];
  const unsigned char* pmask = (const unsigned char*)d_in[21];
  const unsigned char* omask = (const unsigned char*)d_in[22];
  const unsigned char* amask = (const unsigned char*)d_in[23];

  size_t off = 0;
  auto wsp = [&](size_t bytes) -> void* {
    void* p = (char*)d_ws + off;
    off += (bytes + 255) & ~(size_t)255;
    return p;
  };
  const size_t nTok = (size_t)cB * cN;
  float* x    = (float*)wsp(nTok * cD * sizeof(float));
  bf16*  hbuf = (bf16*) wsp(nTok * cD * sizeof(bf16));
  bf16*  qb   = (bf16*) wsp(nTok * cD * sizeof(bf16));
  bf16*  kbuf = (bf16*) wsp(nTok * cD * sizeof(bf16));
  bf16*  vt   = (bf16*) wsp(nTok * cD * sizeof(bf16) + 256);  // +tail slack for padded reads
  bf16*  ob   = (bf16*) wsp(nTok * cD * sizeof(bf16));
  bf16*  m1   = (bf16*) wsp(nTok * cF * sizeof(bf16));
  bf16*  wt   = (bf16*) wsp((size_t)cD * cF * sizeof(bf16));  // transposed bf16 weight scratch

  const int M = cB * cN;                 // 10368 = 81 * 128
  long total = (long)cB * cN * cD;
  k_build<<<dim3((unsigned)((total + 255) / 256)), 256, 0, stream>>>(prefix, obs, act, x);

  dim3 gD((unsigned)(M / 128), cD / 128);   // 81 x 6
  dim3 gF((unsigned)(M / 128), cF / 128);   // 81 x 24
  dim3 gA((unsigned)(cN / 16), cNH, cB);    // 81 x 12 x 8
  dim3 pDD(cD / 32, cD / 64);               // wprep grids (K/32, Nc/64)
  dim3 pDF(cD / 32, cF / 64);
  dim3 pFD(cF / 32, cD / 64);

  for (int l = 0; l < cL; ++l) {
    k_ln<<<M, 256, 0, stream>>>(x, ln1s + (size_t)l * cD, ln1b + (size_t)l * cD, hbuf, nullptr, 0);
    k_wprep<<<pDD, 256, 0, stream>>>(wq + (size_t)l * cD * cD, wt, cD, cD);
    k_gemm<<<gD, 256, 0, stream>>>(hbuf, wt, bq + (size_t)l * cD, nullptr, qb, M, cD, cD, 0);
    k_wprep<<<pDD, 256, 0, stream>>>(wk + (size_t)l * cD * cD, wt, cD, cD);
    k_gemm<<<gD, 256, 0, stream>>>(hbuf, wt, bk + (size_t)l * cD, nullptr, kbuf, M, cD, cD, 0);
    k_wprep<<<pDD, 256, 0, stream>>>(wv + (size_t)l * cD * cD, wt, cD, cD);
    k_gemm<<<gD, 256, 0, stream>>>(hbuf, wt, bv + (size_t)l * cD, nullptr, vt, M, cD, cD, 3);
    k_attn<<<gA, 32, 0, stream>>>(qb, kbuf, vt, pmask, omask, amask, ob);
    k_wprep<<<pDD, 256, 0, stream>>>(wo + (size_t)l * cD * cD, wt, cD, cD);
    k_gemm<<<gD, 256, 0, stream>>>(ob, wt, bo + (size_t)l * cD, x, nullptr, M, cD, cD, 2);
    k_ln<<<M, 256, 0, stream>>>(x, ln2s + (size_t)l * cD, ln2b + (size_t)l * cD, hbuf, nullptr, 0);
    k_wprep<<<pDF, 256, 0, stream>>>(w1 + (size_t)l * cD * cF, wt, cD, cF);
    k_gemm<<<gF, 256, 0, stream>>>(hbuf, wt, b1 + (size_t)l * cF, nullptr, m1, M, cD, cF, 1);
    k_wprep<<<pFD, 256, 0, stream>>>(w2 + (size_t)l * cF * cD, wt, cF, cD);
    k_gemm<<<gD, 256, 0, stream>>>(m1, wt, b2 + (size_t)l * cD, x, nullptr, M, cF, cD, 2);
  }
  k_ln<<<M, 256, 0, stream>>>(x, lnfs, lnfb, nullptr, (float*)d_out, 1);
}
